// NewLSTM_34617436405915
// MI455X (gfx1250) — matrix operand order repaired and checked by reading, not compiled, not run on hardware
//
#include <hip/hip_runtime.h>

// ---------------- problem constants ----------------
#define HDIM 256
#define HH   65536          // 256*256
#define NB   32             // persistent blocks (16 mb-blocks x 2 nb-halves)

// ---------------- workspace layout (bytes) ----------------
#define CNT_OFF  0
#define H_OFF    4096                          // hT ping-pong: 2 * 256*256 * 2B = 262144
#define WX_OFF   (H_OFF + 2 * HH * 2)          // Wx A-fragments: 4*16*8 frags * 1KB = 512KB
#define WH_OFF   (WX_OFF + 524288)             // Wh A-fragments: 512KB
#define XT_OFF   (WH_OFF + 524288)             // XT bf16: steps * 65536 * 2B

typedef __attribute__((ext_vector_type(16))) __bf16 v16bf;
typedef __attribute__((ext_vector_type(8)))  float  v8f;

union FragU { v16bf v; unsigned short s[16]; uint4 u[2]; };

__device__ __forceinline__ unsigned short f2bf(float f) {
    unsigned u = __float_as_uint(f);
    u += 0x7FFFu + ((u >> 16) & 1u);           // round-to-nearest-even
    return (unsigned short)(u >> 16);
}
__device__ __forceinline__ unsigned pack2(float a, float b) {
    return (unsigned)f2bf(a) | ((unsigned)f2bf(b) << 16);
}

__device__ __forceinline__ void gridBarrier(unsigned* cnt, unsigned target) {
    __threadfence();
    __syncthreads();
    if (threadIdx.x == 0) {
        __hip_atomic_fetch_add(cnt, 1u, __ATOMIC_ACQ_REL, __HIP_MEMORY_SCOPE_AGENT);
        while (__hip_atomic_load(cnt, __ATOMIC_ACQUIRE, __HIP_MEMORY_SCOPE_AGENT) < target) {
            __builtin_amdgcn_s_sleep(4);
        }
    }
    __syncthreads();
    __threadfence();
}

// ------------------------------------------------------------------
// Kernel 1: pack Wx=[Wii,Wif,Wig,Wio] and Wh=[Whi,Whf,Whg,Who] rows
// into bf16 WMMA A-fragments, contiguous K (lane half h holds k = 16h..16h+15).  One thread =
// one lane of one 1KB fragment.  Also zeroes the barrier counter.
// ------------------------------------------------------------------
__global__ void wfrag_kernel(const float* __restrict__ Wii, const float* __restrict__ Wif,
                             const float* __restrict__ Wig, const float* __restrict__ Wio,
                             const float* __restrict__ Whi, const float* __restrict__ Whf,
                             const float* __restrict__ Whg, const float* __restrict__ Who,
                             unsigned short* __restrict__ WxA, unsigned short* __restrict__ WhA,
                             unsigned* __restrict__ counter) {
    int gid = blockIdx.x * 256 + threadIdx.x;     // 32768 work items
    if (gid == 0) *counter = 0u;
    int lane = gid & 31;
    int frag = gid >> 5;                          // 0..1023
    int side = frag >> 9;                         // 0 = x-side, 1 = h-side
    int f    = frag & 511;                        // (g*16+mb)*8 + kb
    int kb = f & 7, gm = f >> 3;
    int g = gm >> 4, mb = gm & 15;
    const float* Wx4[4] = {Wii, Wif, Wig, Wio};
    const float* Wh4[4] = {Whi, Whf, Whg, Who};
    const float* W = side ? Wh4[g] : Wx4[g];
    unsigned short* dst = (side ? WhA : WxA) + f * 512 + lane * 16;
    int M = mb * 16 + (lane & 15);
    #pragma unroll
    for (int e = 0; e < 16; ++e) {
        int K = kb * 32 + ((lane >> 4) << 4) + e;
        dst[e] = f2bf(W[M * HDIM + K]);
    }
}

// ------------------------------------------------------------------
// Kernel 2: XT[t][col][k] = bf16(X[t][k][col]) — LDS-tiled transpose.
// ------------------------------------------------------------------
__global__ void xt_kernel(const float* __restrict__ X, unsigned short* __restrict__ XT) {
    __shared__ float ls[32][33];
    int bid  = blockIdx.x;
    int t    = bid >> 6;
    int tile = bid & 63;
    int tr = (tile >> 3) * 32;          // K block
    int tc = (tile & 7) * 32;           // col block
    int tx = threadIdx.x & 31, ty = threadIdx.x >> 5;   // ty 0..7
    const float* src = X + (size_t)t * HH;
    #pragma unroll
    for (int i = 0; i < 4; ++i)
        ls[ty + 8 * i][tx] = src[(size_t)(tr + ty + 8 * i) * HDIM + (tc + tx)];
    __syncthreads();
    unsigned short* dst = XT + (size_t)t * HH;
    #pragma unroll
    for (int i = 0; i < 4; ++i) {
        int c = tc + ty + 8 * i;
        dst[(size_t)c * HDIM + tr + tx] = f2bf(ls[tx][ty + 8 * i]);
    }
}

// ------------------------------------------------------------------
// B-fragment loaders (32x16 bf16, lane half = K half, 16 consecutive K)
// ------------------------------------------------------------------
__device__ __forceinline__ v16bf loadBfrag(const unsigned short* base, int k0) {
    return *(const v16bf*)(const void*)(base + k0);
}
__device__ __forceinline__ v16bf gatherXfrag(const float* xg, int k0) {
    FragU fx;
    const float* xp = xg + (size_t)k0 * HDIM;
    #pragma unroll
    for (int j = 0; j < 16; ++j) fx.s[j] = f2bf(xp[(size_t)j * HDIM]);
    return fx.v;
}

// ------------------------------------------------------------------
// Kernel 3: persistent recurrent LSTM.  32 blocks x 256 threads.
// Block b: mb = b%16, waves own nb = (b/16)*8 + wave; one 16x16 h/c
// tile per wave for all steps (c lives in VGPRs).  Per step: 64 WMMAs
// (4 gates x 8 K-blocks x {Wh,Wx}), fully unrolled; A-frags batched
// from LDS before each WMMA group; B-frags software-pipelined one
// K-block ahead.  Device-wide barrier per step via atomic counter.
// ------------------------------------------------------------------
template <int USE_XT>
__global__ void __launch_bounds__(256) lstm_persistent(
        const unsigned short* __restrict__ XT, const float* __restrict__ X,
        const unsigned short* __restrict__ WhA, const unsigned short* __restrict__ WxA,
        unsigned short* __restrict__ hT,
        const float* __restrict__ bii, const float* __restrict__ bhi,
        const float* __restrict__ bif, const float* __restrict__ bhf,
        const float* __restrict__ bgg, const float* __restrict__ bhg,
        const float* __restrict__ bio, const float* __restrict__ bho,
        float* __restrict__ out, unsigned* __restrict__ counter,
        int steps) {
    __shared__ __attribute__((aligned(32))) unsigned char lds[65536];

    const int lane = threadIdx.x & 31;
    const int wave = threadIdx.x >> 5;
    const int mb   = blockIdx.x & 15;
    const int nb   = (blockIdx.x >> 4) * 8 + wave;
    const int lh   = lane >> 4;               // lane half
    const int ll   = lane & 15;
    const int col  = nb * 16 + ll;            // output column (N)
    const int row0 = mb * 16 + lh * 8;        // first of this lane's 8 rows (M)

    // ---- stage this mb's A-fragments: LDS[0,32K)=Wh, [32K,64K)=Wx ----
    #pragma unroll
    for (int it = 0; it < 16; ++it) {
        int idx  = (int)threadIdx.x + 256 * it;   // uint4 index
        int boff = idx << 4;
        int region = boff >> 15;                  // 0: Wh, 1: Wx
        int rb    = boff & 32767;
        int fragl = rb >> 10;                     // g*8+kb
        int inner = rb & 1023;
        int g = fragl >> 3, kb = fragl & 7;
        const unsigned char* src = (const unsigned char*)(region ? WxA : WhA) +
                                   (((g * 16 + mb) * 8 + kb) * 1024 + inner);
        *(uint4*)(lds + boff) = *(const uint4*)src;
    }

    // ---- constant bias tiles (D layout) ----
    v8f bias[4], c, acc[4];
    const float* bpi[4] = {bii, bif, bgg, bio};
    const float* bph[4] = {bhi, bhf, bhg, bho};
    #pragma unroll
    for (int g = 0; g < 4; ++g)
        #pragma unroll
        for (int r = 0; r < 8; ++r) {
            int a = (row0 + r) * HDIM + col;
            bias[g][r] = bpi[g][a] + bph[g][a];
        }
    #pragma unroll
    for (int r = 0; r < 8; ++r) c[r] = 0.f;

    // ---- h0 = 0 in parity-0 buffer ----
    { uint4 z; z.x = z.y = z.z = z.w = 0u; *(uint4*)(hT + col * HDIM + row0) = z; }
    __syncthreads();
    gridBarrier(counter, NB);

    float hreg[8], oreg[8];
    #pragma unroll
    for (int r = 0; r < 8; ++r) { hreg[r] = 0.f; oreg[r] = 0.f; }

    const int ldsLane = lane * 32;

    for (int t = 0; t < steps; ++t) {
        const unsigned short* hsrc = hT + (t & 1) * HH;
        unsigned short*       hdst = hT + ((t & 1) ^ 1) * HH;
        const unsigned short* hrow = hsrc + col * HDIM;
        const unsigned short* xrow = USE_XT ? (XT + (size_t)t * HH + (size_t)col * HDIM)
                                            : (const unsigned short*)nullptr;
        const float*          xg   = USE_XT ? (const float*)nullptr
                                            : (X + (size_t)t * HH + col);

        acc[0] = bias[0]; acc[1] = bias[1]; acc[2] = bias[2]; acc[3] = bias[3];

        if (USE_XT && (t + 1) < steps)
            __builtin_prefetch(XT + (size_t)(t + 1) * HH + (size_t)col * HDIM, 0, 1);

        // ---- software-pipelined B fragments, batched A fragments ----
        v16bf bh, bx;
        {
            int k0 = lh * 16;
            bh = loadBfrag(hrow, k0);
            bx = USE_XT ? loadBfrag(xrow, k0) : gatherXfrag(xg, k0);
        }
        #pragma unroll
        for (int kb = 0; kb < 8; ++kb) {
            v16bf bh_n, bx_n;
            if (kb < 7) {
                int k0n = (kb + 1) * 32 + lh * 16;
                bh_n = loadBfrag(hrow, k0n);
                bx_n = USE_XT ? loadBfrag(xrow, k0n) : gatherXfrag(xg, k0n);
            }
            // batch all 8 A-fragments for this K-block from LDS
            v16bf ah[4], ax[4];
            #pragma unroll
            for (int g = 0; g < 4; ++g) {
                ah[g] = *(const v16bf*)(const void*)(lds + (g * 8 + kb) * 1024 + ldsLane);
                ax[g] = *(const v16bf*)(const void*)(lds + 32768 + (g * 8 + kb) * 1024 + ldsLane);
            }
            // 8 back-to-back WMMAs on 4 independent accumulator chains
            #pragma unroll
            for (int g = 0; g < 4; ++g)
                acc[g] = __builtin_amdgcn_wmma_f32_16x16x32_bf16(false, ah[g], false, bh,
                                                                 (short)0, acc[g], false, false);
            #pragma unroll
            for (int g = 0; g < 4; ++g)
                acc[g] = __builtin_amdgcn_wmma_f32_16x16x32_bf16(false, ax[g], false, bx,
                                                                 (short)0, acc[g], false, false);
            if (kb < 7) { bh = bh_n; bx = bx_n; }
        }

        // ---- gates + state update (fast sigmoid/tanh via v_exp_f32) ----
        #pragma unroll
        for (int r = 0; r < 8; ++r) {
            float ig = 1.f / (1.f + __expf(-acc[0][r]));
            float fg = 1.f / (1.f + __expf(-acc[1][r]));
            float gg = 1.f - 2.f / (__expf(2.f * acc[2][r]) + 1.f);
            float og = 1.f / (1.f + __expf(-acc[3][r]));
            float cn = fg * c[r] + ig * gg;
            c[r] = cn;
            hreg[r] = og * (1.f - 2.f / (__expf(2.f * cn) + 1.f));
            oreg[r] = og;
        }

        // ---- store h_new transposed bf16: one 16B store per lane ----
        uint4 hp;
        hp.x = pack2(hreg[0], hreg[1]);
        hp.y = pack2(hreg[2], hreg[3]);
        hp.z = pack2(hreg[4], hreg[5]);
        hp.w = pack2(hreg[6], hreg[7]);
        *(uint4*)(hdst + col * HDIM + row0) = hp;

        gridBarrier(counter, (unsigned)NB * (unsigned)(t + 2));
    }

    // ---- outputs: concat(o, h, c), fp32 row-major ----
    #pragma unroll
    for (int r = 0; r < 8; ++r) {
        int a = (row0 + r) * HDIM + col;
        out[a]           = oreg[r];
        out[HH + a]      = hreg[r];
        out[2 * HH + a]  = c[r];
    }
}

// ------------------------------------------------------------------
extern "C" void kernel_launch(void* const* d_in, const int* in_sizes, int n_in,
                              void* d_out, int out_size, void* d_ws, size_t ws_size,
                              hipStream_t stream) {
    const float* X   = (const float*)d_in[0];
    const float* Wii = (const float*)d_in[1];
    const float* Whi = (const float*)d_in[2];
    const float* Wif = (const float*)d_in[3];
    const float* Whf = (const float*)d_in[4];
    const float* Wig = (const float*)d_in[5];
    const float* Whg = (const float*)d_in[6];
    const float* Wio = (const float*)d_in[7];
    const float* Who = (const float*)d_in[8];
    const float* bii = (const float*)d_in[9];
    const float* bhi = (const float*)d_in[10];
    const float* bif = (const float*)d_in[11];
    const float* bhf = (const float*)d_in[12];
    const float* bgg = (const float*)d_in[13];
    const float* bhg = (const float*)d_in[14];
    const float* bio = (const float*)d_in[15];
    const float* bho = (const float*)d_in[16];

    int T = in_sizes[0] / HH;
    int steps = T - 1;
    if (steps < 0) steps = 0;

    char* ws = (char*)d_ws;
    if (ws_size < (size_t)XT_OFF) return;   // cannot run without minimal scratch

    unsigned*       cnt = (unsigned*)(ws + CNT_OFF);
    unsigned short* hT  = (unsigned short*)(ws + H_OFF);
    unsigned short* WxA = (unsigned short*)(ws + WX_OFF);
    unsigned short* WhA = (unsigned short*)(ws + WH_OFF);
    unsigned short* XT  = (unsigned short*)(ws + XT_OFF);

    size_t xt_bytes = (size_t)steps * HH * 2;
    int use_xt = (ws_size >= (size_t)XT_OFF + xt_bytes) ? 1 : 0;

    wfrag_kernel<<<128, 256, 0, stream>>>(Wii, Wif, Wig, Wio, Whi, Whf, Whg, Who,
                                          WxA, WhA, cnt);
    if (use_xt && steps > 0)
        xt_kernel<<<steps * 64, 256, 0, stream>>>(X, XT);

    if (use_xt)
        lstm_persistent<1><<<NB, 256, 0, stream>>>(XT, X, WhA, WxA, hT,
                                                   bii, bhi, bif, bhf, bgg, bhg, bio, bho,
                                                   (float*)d_out, cnt, steps);
    else
        lstm_persistent<0><<<NB, 256, 0, stream>>>(XT, X, WhA, WxA, hT,
                                                   bii, bhi, bif, bhf, bgg, bhg, bio, bho,
                                                   (float*)d_out, cnt, steps);
}
